// Autocorrelation_83502754169522
// MI455X (gfx1250) — matrix operand order repaired and checked by reading, not compiled, not run on hardware
//
#include <hip/hip_runtime.h>
#include <hip/hip_bf16.h>

typedef __attribute__((ext_vector_type(16))) __bf16 v16bf;
typedef __attribute__((ext_vector_type(8)))  __bf16 v8bf;
typedef __attribute__((ext_vector_type(8)))  float  v8f;

#define B_    4
#define L_    2048
#define DM    1024
#define DH    64
#define KTOP  15
#define ROWS  (B_ * DH)     /* 256 rows of [L] in q/k/v transposed layout */
#define MROWS (B_ * L_)     /* 8192 rows for the projection GEMM          */

#if __has_builtin(__builtin_amdgcn_global_load_async_to_lds_b128) && \
    __has_builtin(__builtin_amdgcn_s_wait_asynccnt)
#define USE_ASYNC_LDS 1
typedef int v4i __attribute__((vector_size(16)));
typedef __attribute__((address_space(1))) v4i as1_v4i;   // global (printed __device__)
typedef __attribute__((address_space(3))) v4i as3_v4i;   // LDS    (printed __shared__)
static __device__ __forceinline__ void async_copy_b128(const float* g, float* l) {
  __builtin_amdgcn_global_load_async_to_lds_b128(
      (as1_v4i*)g, (as3_v4i*)l, 0, 0);
}
#else
#define USE_ASYNC_LDS 0
#endif

// ---------------------------------------------------------------------------
// Prologue: convert + transpose weights to bf16, n-major (so a WMMA B-fragment
// lane load is one contiguous 32B v16bf load).
// WqT  [DH][DM], WoutT [DM][DM]
// ---------------------------------------------------------------------------
__global__ __launch_bounds__(256) void prep_weights(const float* __restrict__ Wq,
                                                    const float* __restrict__ Wout,
                                                    __bf16* __restrict__ WqT,
                                                    __bf16* __restrict__ WoutT) {
  const int total = DH * DM + DM * DM;
  for (int i = blockIdx.x * blockDim.x + threadIdx.x; i < total;
       i += gridDim.x * blockDim.x) {
    if (i < DH * DM) {
      int n = i / DM, k = i - n * DM;
      WqT[i] = (__bf16)Wq[k * DH + n];
    } else {
      int j = i - DH * DM;
      int n = j / DM, k = j - n * DM;
      WoutT[j] = (__bf16)Wout[k * DM + n];
    }
  }
}

// ---------------------------------------------------------------------------
// Projection GEMM: [8192,1024]f32 x [1024,64]bf16 (+bq) for Q,K,V.
// One wave per 16x64 strip, 4 f32 accumulators, K stepped by 32, with
// software-pipelined (double-buffered) A/B fragments; the loop is unrolled
// by 2 so the buffer rotation is pure register renaming (no v_mov chains).
// ---------------------------------------------------------------------------
__global__ __launch_bounds__(128) void proj_wmma(const float* __restrict__ Q,
                                                 const float* __restrict__ Kx,
                                                 const float* __restrict__ V,
                                                 const __bf16* __restrict__ WqT,
                                                 const float* __restrict__ bq,
                                                 float* __restrict__ qT,
                                                 float* __restrict__ kT,
                                                 float* __restrict__ vT) {
  const int which = blockIdx.y;
  const float* X = (which == 0) ? Q : (which == 1) ? Kx : V;
  float*       O = (which == 0) ? qT : (which == 1) ? kT : vT;

  const int wid  = threadIdx.x >> 5;
  const int lane = threadIdx.x & 31;
  const int khi  = lane >> 4;      // which K half-row this lane holds
  const int lm   = lane & 15;      // M row (A) / N col (B,C)
  const int m0   = (blockIdx.x * 4 + wid) * 16;
  const float* pr = X + (size_t)(m0 + lm) * DM;

  v8f acc[4] = {};

  float4 f0 = *(const float4*)(pr + khi * 16);
  float4 f1 = *(const float4*)(pr + khi * 16 + 4);
  float4 f2 = *(const float4*)(pr + khi * 16 + 8);
  float4 f3 = *(const float4*)(pr + khi * 16 + 12);
  v16bf bfr[4];
#pragma unroll
  for (int nt = 0; nt < 4; nt++)
    bfr[nt] = *(const v16bf*)(WqT + (size_t)(nt * 16 + lm) * DM + khi * 16);

#pragma unroll 2
  for (int kk = 0; kk < DM; kk += 32) {
    // prefetch next K-step fragments (stay resident while WMMAs run)
    float4 g0 = f0, g1 = f1, g2 = f2, g3 = f3;
    v16bf  bn[4] = {bfr[0], bfr[1], bfr[2], bfr[3]};
    if (kk + 32 < DM) {
      const int kn = kk + 32;
      g0 = *(const float4*)(pr + kn + khi * 16);
      g1 = *(const float4*)(pr + kn + khi * 16 + 4);
      g2 = *(const float4*)(pr + kn + khi * 16 + 8);
      g3 = *(const float4*)(pr + kn + khi * 16 + 12);
#pragma unroll
      for (int nt = 0; nt < 4; nt++)
        bn[nt] = *(const v16bf*)(WqT + (size_t)(nt * 16 + lm) * DM + kn + khi * 16);
    }
    // A fragment (16x32 bf16): element j -> K = khi*16 + j
    v16bf a;
    a[0]=(__bf16)f0.x;  a[1]=(__bf16)f0.y;  a[2]=(__bf16)f0.z;  a[3]=(__bf16)f0.w;
    a[4]=(__bf16)f1.x;  a[5]=(__bf16)f1.y;  a[6]=(__bf16)f1.z;  a[7]=(__bf16)f1.w;
    a[8]=(__bf16)f2.x;  a[9]=(__bf16)f2.y;  a[10]=(__bf16)f2.z; a[11]=(__bf16)f2.w;
    a[12]=(__bf16)f3.x; a[13]=(__bf16)f3.y; a[14]=(__bf16)f3.z; a[15]=(__bf16)f3.w;

#pragma unroll
    for (int nt = 0; nt < 4; nt++)
      acc[nt] = __builtin_amdgcn_wmma_f32_16x16x32_bf16(
          false, a, false, bfr[nt], (short)0, acc[nt], false, false);

    f0 = g0; f1 = g1; f2 = g2; f3 = g3;
#pragma unroll
    for (int nt = 0; nt < 4; nt++) bfr[nt] = bn[nt];
  }

  // Epilogue: +bq, scatter into [B, DH, L]
#pragma unroll
  for (int nt = 0; nt < 4; nt++) {
    const int d = nt * 16 + lm;
    const float bias = bq[d];
#pragma unroll
    for (int r = 0; r < 8; r++) {
      const int m = m0 + khi * 8 + r;
      const int b = m >> 11, l = m & (L_ - 1);
      O[((size_t)(b * DH + d)) * L_ + l] = acc[nt][r] + bias;
    }
  }
}

// ---------------------------------------------------------------------------
// Circular cross-correlation, replacing the FFT path:
//   corr[tau] = (1/256) * sum_{r,t} q[r][(t+tau)%L] * k[r][t]
// 16 tau per block.  Rows double-buffered in LDS; staged with
// GLOBAL_LOAD_ASYNC_TO_LDS_B128 (ASYNCcnt) when the toolchain exposes it:
// issue row r+1, s_wait_asynccnt 4 (in-order) to guarantee row r landed,
// barrier, compute.
// ---------------------------------------------------------------------------
__global__ __launch_bounds__(256) void corr_kernel(const float* __restrict__ qT,
                                                   const float* __restrict__ kT,
                                                   float* __restrict__ corr) {
  __shared__ float qs[2][L_];
  __shared__ float ks[2][L_];
  const int tid  = threadIdx.x;
  const int tau0 = blockIdx.x * 16;

  float acc[16] = {};

#if USE_ASYNC_LDS
  {
    const float* qr = qT;  // row 0
    const float* kr = kT;
    async_copy_b128(qr + tid * 4,         &qs[0][tid * 4]);
    async_copy_b128(qr + (tid + 256) * 4, &qs[0][(tid + 256) * 4]);
    async_copy_b128(kr + tid * 4,         &ks[0][tid * 4]);
    async_copy_b128(kr + (tid + 256) * 4, &ks[0][(tid + 256) * 4]);
  }
#endif

  for (int r = 0; r < ROWS; r++) {
    const int cur = r & 1;
#if USE_ASYNC_LDS
    if (r + 1 < ROWS) {                       // stage next row while computing
      const int nxt = cur ^ 1;
      const float* qr = qT + (size_t)(r + 1) * L_;
      const float* kr = kT + (size_t)(r + 1) * L_;
      async_copy_b128(qr + tid * 4,         &qs[nxt][tid * 4]);
      async_copy_b128(qr + (tid + 256) * 4, &qs[nxt][(tid + 256) * 4]);
      async_copy_b128(kr + tid * 4,         &ks[nxt][tid * 4]);
      async_copy_b128(kr + (tid + 256) * 4, &ks[nxt][(tid + 256) * 4]);
      __builtin_amdgcn_s_wait_asynccnt(4);    // row r's 4 copies complete
    } else {
      __builtin_amdgcn_s_wait_asynccnt(0);
    }
    __syncthreads();
#else
    const float* qr = qT + (size_t)r * L_;
    const float* kr = kT + (size_t)r * L_;
    __syncthreads();
#pragma unroll
    for (int i = 0; i < 8; i++) {
      qs[cur][tid + i * 256] = qr[tid + i * 256];
      ks[cur][tid + i * 256] = kr[tid + i * 256];
    }
    if (r + 1 < ROWS) {                       // global_prefetch_b8
      __builtin_prefetch(qr + L_, 0, 0);
      __builtin_prefetch(kr + L_, 0, 0);
    }
    __syncthreads();
#endif
#pragma unroll
    for (int i = 0; i < 8; i++) {
      const int t = tid + i * 256;
      const float kv = ks[cur][t];
#pragma unroll
      for (int j = 0; j < 16; j++)
        acc[j] += qs[cur][(t + tau0 + j) & (L_ - 1)] * kv;
    }
    __syncthreads();                          // done reading before next stage
  }

  float* red = &qs[0][0];
  for (int j = 0; j < 16; j++) {
    __syncthreads();
    red[tid] = acc[j];
    __syncthreads();
    for (int s = 128; s > 0; s >>= 1) {
      if (tid < s) red[tid] += red[tid + s];
      __syncthreads();
    }
    if (tid == 0) corr[tau0 + j] = red[0] * (1.0f / (float)ROWS);
  }
}

// ---------------------------------------------------------------------------
// Top-15 (lowest-index tiebreak, matching top_k) + softmax weights.
// ---------------------------------------------------------------------------
__global__ __launch_bounds__(256) void topk_kernel(const float* __restrict__ corr,
                                                   float* __restrict__ w,
                                                   int* __restrict__ sh) {
  __shared__ float v[L_];
  __shared__ float rv[256];
  __shared__ int   ri[256];
  __shared__ float tv[KTOP];
  __shared__ int   ti[KTOP];
  const int tid = threadIdx.x;
  for (int i = 0; i < 8; i++) v[tid + i * 256] = corr[tid + i * 256];
  __syncthreads();

  for (int it = 0; it < KTOP; it++) {
    float best = -3.0e38f; int bi = 0;
    for (int i = 0; i < 8; i++) {
      const int idx = tid + i * 256;
      const float x = v[idx];
      if (x > best) { best = x; bi = idx; }
    }
    rv[tid] = best; ri[tid] = bi;
    __syncthreads();
    for (int s = 128; s > 0; s >>= 1) {
      if (tid < s) {
        if (rv[tid + s] > rv[tid] ||
            (rv[tid + s] == rv[tid] && ri[tid + s] < ri[tid])) {
          rv[tid] = rv[tid + s]; ri[tid] = ri[tid + s];
        }
      }
      __syncthreads();
    }
    if (tid == 0) { tv[it] = rv[0]; ti[it] = ri[0]; v[ri[0]] = -3.0e38f; }
    __syncthreads();
  }

  if (tid == 0) {
    float m = tv[0];
    for (int i = 1; i < KTOP; i++) m = fmaxf(m, tv[i]);
    float e[KTOP], s = 0.f;
    for (int i = 0; i < KTOP; i++) { e[i] = __expf(tv[i] - m); s += e[i]; }
    const float inv = 1.0f / s;
    for (int i = 0; i < KTOP; i++) { w[i] = e[i] * inv; sh[i] = ti[i]; }
  }
}

// ---------------------------------------------------------------------------
// R3 = sum_i w_i * roll(v, -shift_i); stored directly as the bf16 A-matrix
// [512,1024] of the final GEMM (the buggy reshape makes them bit-identical).
// ---------------------------------------------------------------------------
__global__ __launch_bounds__(256) void delay_kernel(const float* __restrict__ vT,
                                                    const float* __restrict__ w,
                                                    const int* __restrict__ sh,
                                                    __bf16* __restrict__ A) {
  __shared__ float ws_[KTOP];
  __shared__ int   ss_[KTOP];
  if (threadIdx.x < KTOP) { ws_[threadIdx.x] = w[threadIdx.x]; ss_[threadIdx.x] = sh[threadIdx.x]; }
  __syncthreads();
  const int e = blockIdx.x * 256 + threadIdx.x;   // 524288 elements exactly
  const int r = e >> 11, t = e & (L_ - 1);
  const float* row = vT + (size_t)r * L_;
  float acc = 0.f;
#pragma unroll
  for (int i = 0; i < KTOP; i++) acc += ws_[i] * row[(t + ss_[i]) & (L_ - 1)];
  A[e] = (__bf16)acc;
}

// ---------------------------------------------------------------------------
// Collapsed output GEMM: C[512,1024] = A(bf16) x Wout (via WoutT bf16) + bout.
// 8 waves/block: 2 M-tiles x 4 wave-columns -> block covers 32x256.
// Double-buffered fragments, unrolled by 2 for mov-free rotation.
// ---------------------------------------------------------------------------
__global__ __launch_bounds__(256) void out_gemm(const __bf16* __restrict__ A,
                                                const __bf16* __restrict__ WoutT,
                                                const float* __restrict__ bout,
                                                float* __restrict__ C) {
  const int wid  = threadIdx.x >> 5;
  const int lane = threadIdx.x & 31;
  const int khi  = lane >> 4;
  const int lm   = lane & 15;
  const int m0   = blockIdx.y * 32 + (wid >> 2) * 16;
  const int n0   = blockIdx.x * 256 + (wid & 3) * 64;

  v8f acc[4] = {};
  const __bf16* arow = A + (size_t)(m0 + lm) * DM;

  v8bf a0 = *(const v8bf*)(arow + khi * 16);
  v8bf a1 = *(const v8bf*)(arow + khi * 16 + 8);
  v16bf bfr[4];
#pragma unroll
  for (int nt = 0; nt < 4; nt++)
    bfr[nt] = *(const v16bf*)(WoutT + (size_t)(n0 + nt * 16 + lm) * DM + khi * 16);

#pragma unroll 2
  for (int kk = 0; kk < DM; kk += 32) {
    v8bf g0 = a0, g1 = a1;
    v16bf bn[4] = {bfr[0], bfr[1], bfr[2], bfr[3]};
    if (kk + 32 < DM) {
      const int kn = kk + 32;
      g0 = *(const v8bf*)(arow + kn + khi * 16);
      g1 = *(const v8bf*)(arow + kn + khi * 16 + 8);
#pragma unroll
      for (int nt = 0; nt < 4; nt++)
        bn[nt] = *(const v16bf*)(WoutT + (size_t)(n0 + nt * 16 + lm) * DM + kn + khi * 16);
    }
    v16bf a = __builtin_shufflevector(a0, a1, 0,1,2,3,4,5,6,7,8,9,10,11,12,13,14,15);
#pragma unroll
    for (int nt = 0; nt < 4; nt++)
      acc[nt] = __builtin_amdgcn_wmma_f32_16x16x32_bf16(
          false, a, false, bfr[nt], (short)0, acc[nt], false, false);
    a0 = g0; a1 = g1;
#pragma unroll
    for (int nt = 0; nt < 4; nt++) bfr[nt] = bn[nt];
  }
#pragma unroll
  for (int nt = 0; nt < 4; nt++) {
    const int n = n0 + nt * 16 + lm;
    const float bias = bout[n];
#pragma unroll
    for (int r = 0; r < 8; r++) {
      const int mm = m0 + khi * 8 + r;
      C[(size_t)mm * DM + n] = acc[nt][r] + bias;
    }
  }
}

// ---------------------------------------------------------------------------
// Broadcast the 512 distinct rows to the full [4,2048,1024] output.
// row(l) = b*128 + ((l>>1)&63)*2 + (l&1)
// ---------------------------------------------------------------------------
__global__ __launch_bounds__(256) void scatter_out(const float* __restrict__ C,
                                                   float* __restrict__ out) {
  const int g = blockIdx.x * 256 + threadIdx.x;  // 2,097,152 float4s
  const int n4 = g & 255;
  const int rowIdx = g >> 8;                     // [0, 8192)
  const int b = rowIdx >> 11, l = rowIdx & (L_ - 1);
  const int crow = b * 128 + ((l >> 1) & 63) * 2 + (l & 1);
  float4 val = *(const float4*)(C + (size_t)crow * DM + n4 * 4);
  *(float4*)(out + (size_t)rowIdx * DM + n4 * 4) = val;
}

// ---------------------------------------------------------------------------
extern "C" void kernel_launch(void* const* d_in, const int* in_sizes, int n_in,
                              void* d_out, int out_size, void* d_ws, size_t ws_size,
                              hipStream_t stream) {
  (void)in_sizes; (void)n_in; (void)out_size; (void)ws_size;
  const float* Q    = (const float*)d_in[0];
  const float* K    = (const float*)d_in[1];
  const float* V    = (const float*)d_in[2];
  const float* Wq   = (const float*)d_in[3];
  const float* bq   = (const float*)d_in[4];
  const float* Wout = (const float*)d_in[5];
  const float* bout = (const float*)d_in[6];
  float* out = (float*)d_out;

  char* ws = (char*)d_ws;                       // ~11.7 MB used
  float*  qT    = (float*)(ws);                                  // 2 MB
  float*  kT    = (float*)(ws + (size_t)(2u << 20));             // 2 MB
  float*  vT    = (float*)(ws + (size_t)(4u << 20));             // 2 MB
  __bf16* WqT   = (__bf16*)(ws + (size_t)(6u << 20));            // 128 KB
  __bf16* WoutT = (__bf16*)(ws + (size_t)(6u << 20) + (128u << 10)); // 2 MB
  float*  corr  = (float*)(ws + (size_t)(8u << 20) + (128u << 10));  // 8 KB
  float*  wv    = (float*)(ws + (size_t)(8u << 20) + (136u << 10));  // 1 KB
  int*    sh    = (int*)  (ws + (size_t)(8u << 20) + (137u << 10));  // 1 KB
  __bf16* Abf   = (__bf16*)(ws + (size_t)(8u << 20) + (138u << 10)); // 1 MB
  float*  Cm    = (float*)(ws + (size_t)(9u << 20) + (138u << 10));  // 2 MB

  prep_weights<<<dim3(2048), 256, 0, stream>>>(Wq, Wout, WqT, WoutT);
  proj_wmma<<<dim3(128, 3), 128, 0, stream>>>(Q, K, V, WqT, bq, qT, kT, vT);
  corr_kernel<<<dim3(128), 256, 0, stream>>>(qT, kT, corr);
  topk_kernel<<<dim3(1), 256, 0, stream>>>(corr, wv, sh);
  delay_kernel<<<dim3(2048), 256, 0, stream>>>(vT, wv, sh, Abf);
  out_gemm<<<dim3(4, 16), 256, 0, stream>>>(Abf, WoutT, bout, Cm);
  scatter_out<<<dim3(8192), 256, 0, stream>>>(Cm, out);
}
